// MultiHeadAttention_56367150792919
// MI455X (gfx1250) — compile-verified, matrix operand order checked by reading, not run on hardware
//
#include <hip/hip_runtime.h>

// ---------------------------------------------------------------------------
// MHA forward for MI455X (gfx1250): bf16 WMMA, fp32 accumulate, async LDS copy.
//   x:[2,2048,1024] qkv_w:[3072,1024] qkv_b:[3072] out_w:[1024,1024] out_b:[1024]
// Pipeline: cvt(f32->bf16) -> GEMM qkv -> flash attention -> GEMM out-proj
// ---------------------------------------------------------------------------

typedef __bf16 bf16_t;
typedef __attribute__((ext_vector_type(16))) __bf16 v16bf;
typedef __attribute__((ext_vector_type(8)))  float  v8f;

#define D_MODEL  1024
#define N_HEADS  16
#define BATCH    2
#define SEQ      2048
#define HEAD_DIM 64
#define M_ROWS   (BATCH * SEQ)     // 4096

// Assemble a 16x32 bf16 WMMA fragment (A-layout or symmetric B-layout) from a
// K-contiguous row: elements 0..7 = K[khalf..khalf+7], 8..15 = K[khalf+16..+23].
static __device__ __forceinline__ v16bf load_frag16(const bf16_t* p) {
  union { v16bf v; uint4 q[2]; } u;
  u.q[0] = *(const uint4*)(p);
  u.q[1] = *(const uint4*)(p + 16);
  return u.v;
}

static __device__ __forceinline__ v8f wmma_bf16(v16bf a, v16bf b, v8f c) {
  return __builtin_amdgcn_wmma_f32_16x16x32_bf16(false, a, false, b,
                                                 (short)0, c, false, false);
}

// CDNA5 async global->LDS copy (ASYNCcnt-tracked, bypasses VGPRs).
// Generic shared pointers carry the LDS byte offset in their low 32 bits.
static __device__ __forceinline__ void async_ld_b128(void* lds, const void* g) {
  unsigned l = (unsigned)(size_t)lds;
  asm volatile("global_load_async_to_lds_b128 %0, %1, off"
               :: "v"(l), "v"(g) : "memory");
}
static __device__ __forceinline__ void wait_async() {
  asm volatile("s_wait_asynccnt 0" ::: "memory");
}

// VALU xor-shuffle within 16-lane groups via v_permlane16_b32 (no LDS traffic).
constexpr unsigned sel_lo(int m) {
  unsigned s = 0;
  for (int i = 0; i < 8; ++i) s |= (unsigned)((i ^ m) & 15) << (4 * i);
  return s;
}
constexpr unsigned sel_hi(int m) {
  unsigned s = 0;
  for (int i = 8; i < 16; ++i) s |= (unsigned)((i ^ m) & 15) << (4 * (i - 8));
  return s;
}
template<int M>
static __device__ __forceinline__ float permxor16(float v) {
  constexpr unsigned s0 = sel_lo(M), s1 = sel_hi(M);
  int x = __builtin_bit_cast(int, v);
  int y = __builtin_amdgcn_permlane16(x, x, (int)s0, (int)s1, false, false);
  return __builtin_bit_cast(float, y);
}

// ---------------------------------------------------------------------------
// fp32 -> bf16 conversion (n multiple of 4)
// ---------------------------------------------------------------------------
__global__ __launch_bounds__(256)
void cvt_f32_bf16(const float* __restrict__ in, bf16_t* __restrict__ out, int n) {
  int i = (blockIdx.x * blockDim.x + threadIdx.x) * 4;
  if (i < n) {
    float4 f = *(const float4*)(in + i);
    union { ushort4 u; bf16_t e[4]; } o;
    o.e[0] = (bf16_t)f.x; o.e[1] = (bf16_t)f.y;
    o.e[2] = (bf16_t)f.z; o.e[3] = (bf16_t)f.w;
    *(ushort4*)(out + i) = o.u;
  }
}

// ---------------------------------------------------------------------------
// C[M,N] = A[M,K] @ B[N,K]^T + bias[N]   (M%128==N%128==K%32==0, all tiles full)
// Block: 256 threads = 8 waves (4x2); wave owns 32x64 (2x4 WMMA tiles).
// Double-buffered LDS, filled by async global->LDS b128 copies; tile t+1 is
// prefetched while WMMAs consume tile t. LDS rows padded to 40 elems (80 B)
// -> conflict-free b128 fragment reads. Total LDS: 4 * 10 KB = 40 KB.
// ---------------------------------------------------------------------------
template<bool OUT_BF16>
__global__ __launch_bounds__(256)
void gemm_bf16_tn(const bf16_t* __restrict__ A,
                  const bf16_t* __restrict__ B,
                  const float*  __restrict__ bias,
                  void* __restrict__ Cout,
                  int M, int N, int K)
{
  __shared__ __align__(16) bf16_t As[2][128 * 40];
  __shared__ __align__(16) bf16_t Bs[2][128 * 40];

  const int tid   = threadIdx.x;
  const int lane  = tid & 31;
  const int wave  = tid >> 5;
  const int wm    = wave >> 1;              // 0..3 -> 32-row strip
  const int wn    = wave & 1;               // 0..1 -> 64-col strip
  const int lrow  = lane & 15;
  const int khalf = (lane < 16) ? 0 : 8;
  const int rhalf = (lane < 16) ? 0 : 8;
  const int bm = blockIdx.y * 128;
  const int bn = blockIdx.x * 128;

  v8f acc[2][4];
#pragma unroll
  for (int i = 0; i < 2; ++i)
#pragma unroll
    for (int j = 0; j < 4; ++j) { v8f z = {}; acc[i][j] = z; }

  auto stage = [&](int buf, int k0) {
#pragma unroll
    for (int t = 0; t < 2; ++t) {
      int i  = tid + t * 256;        // 0..511 : 128 rows x 4 b128 chunks
      int r  = i >> 2;
      int kc = (i & 3) * 8;
      async_ld_b128(&As[buf][r * 40 + kc], &A[(size_t)(bm + r) * K + k0 + kc]);
      async_ld_b128(&Bs[buf][r * 40 + kc], &B[(size_t)(bn + r) * K + k0 + kc]);
    }
  };

  stage(0, 0);
  wait_async();
  __syncthreads();

  int buf = 0;
  for (int k0 = 0; k0 < K; k0 += 32, buf ^= 1) {
    if (k0 + 32 < K) stage(buf ^ 1, k0 + 32);   // prefetch next tile (async)

    v16bf af[2], bfr[4];
#pragma unroll
    for (int i = 0; i < 2; ++i)
      af[i] = load_frag16(&As[buf][(wm * 32 + i * 16 + lrow) * 40 + khalf]);
#pragma unroll
    for (int j = 0; j < 4; ++j)
      bfr[j] = load_frag16(&Bs[buf][(wn * 64 + j * 16 + lrow) * 40 + khalf]);
#pragma unroll
    for (int i = 0; i < 2; ++i)
#pragma unroll
      for (int j = 0; j < 4; ++j)
        acc[i][j] = wmma_bf16(af[i], bfr[j], acc[i][j]);

    wait_async();      // own async ops into next buffer complete
    __syncthreads();   // everyone's complete -> safe to consume / re-stage
  }

  // epilogue: C layout -> VGPR r holds row (r + rhalf), lane holds col lrow
#pragma unroll
  for (int i = 0; i < 2; ++i) {
    const int rbase = bm + wm * 32 + i * 16 + rhalf;
#pragma unroll
    for (int j = 0; j < 4; ++j) {
      const int gcol = bn + wn * 64 + j * 16 + lrow;
      const float bv = bias[gcol];
#pragma unroll
      for (int r = 0; r < 8; ++r) {
        float v = acc[i][j][r] + bv;
        size_t idx = (size_t)(rbase + r) * N + gcol;
        if (OUT_BF16) ((bf16_t*)Cout)[idx] = (bf16_t)v;
        else          ((float*)Cout)[idx]  = v;
      }
    }
  }
}

// ---------------------------------------------------------------------------
// Flash attention per (batch, head, 64-row i-block). Reference quirk:
// scores[i][j] = dot(K_i, Q_j) / sqrt(64); softmax over j; O = alpha @ V.
// Block: 128 threads = 4 waves; wave w owns score rows i0+16w .. +15.
// ---------------------------------------------------------------------------
__global__ __launch_bounds__(128)
void attention_kernel(const bf16_t* __restrict__ qkv,   // [B*S, 3*D]
                      bf16_t* __restrict__ attn)        // [B*S, D]
{
  __shared__ __align__(16) bf16_t Qs [64 * 72];     // j x k  ("keys" = Q rows)
  __shared__ __align__(16) bf16_t Vts[64 * 72];     // c x j  (V transposed)
  __shared__ __align__(16) bf16_t Ps [4 * 16 * 72]; // per-wave P staging

  const int tid   = threadIdx.x;
  const int lane  = tid & 31;
  const int wave  = tid >> 5;          // 0..3
  const int lrow  = lane & 15;
  const int khalf = (lane < 16) ? 0 : 8;
  const int rhalf = (lane < 16) ? 0 : 8;

  const int bh = blockIdx.y;
  const int b  = bh / N_HEADS;
  const int h  = bh % N_HEADS;
  const int i0 = blockIdx.x * 64;

  const size_t rowstride = 3 * D_MODEL;
  const size_t base      = (size_t)(b * SEQ) * rowstride;
  const int    hoff      = h * HEAD_DIM;

  // A fragments: K-matrix rows (acting as queries), loaded once to registers
  v16bf afrag[2];
#pragma unroll
  for (int ks = 0; ks < 2; ++ks)
    afrag[ks] = load_frag16(qkv + base + (size_t)(i0 + wave * 16 + lrow) * rowstride
                            + D_MODEL + hoff + ks * 32 + khalf);

  float m_i[8], l_i[8];
  v8f oacc[4];
#pragma unroll
  for (int r = 0; r < 8; ++r) { m_i[r] = -1e30f; l_i[r] = 0.0f; }
#pragma unroll
  for (int n = 0; n < 4; ++n) { v8f z = {}; oacc[n] = z; }

  bf16_t* pw = &Ps[wave * 16 * 72];

  for (int jt = 0; jt < SEQ / 64; ++jt) {
    const int j0 = jt * 64;
    __syncthreads();   // previous iteration's readers are done
    // stage Q tile (async b128, row-major) and V tile (transposed, manual)
#pragma unroll
    for (int t = 0; t < 4; ++t) {
      int i  = tid + t * 128;          // 0..511
      int r  = i >> 3;                 // j row 0..63
      int c8 = (i & 7) * 8;
      async_ld_b128(&Qs[r * 72 + c8],
                    &qkv[base + (size_t)(j0 + r) * rowstride + hoff + c8]);
      union { uint4 q; bf16_t e[8]; } u;
      u.q = *(const uint4*)&qkv[base + (size_t)(j0 + r) * rowstride + 2 * D_MODEL + hoff + c8];
#pragma unroll
      for (int e = 0; e < 8; ++e)
        Vts[(c8 + e) * 72 + r] = u.e[e];
    }
    wait_async();
    __syncthreads();

    // S = K_rows @ Q_rows^T  (16 x 64 per wave), K-dim = 64 -> 2 WMMA k-steps
    v8f sacc[4];
#pragma unroll
    for (int n = 0; n < 4; ++n) { v8f z = {}; sacc[n] = z; }
#pragma unroll
    for (int ks = 0; ks < 2; ++ks)
#pragma unroll
      for (int n = 0; n < 4; ++n) {
        v16bf bfq = load_frag16(&Qs[(n * 16 + lrow) * 72 + ks * 32 + khalf]);
        sacc[n] = wmma_bf16(afrag[ks], bfq, sacc[n]);
      }

    // scale + online softmax; row stats reduced across each 16-lane half with
    // v_permlane16 (pure VALU, no LDS traffic)
#pragma unroll
    for (int r = 0; r < 8; ++r) {
      float mx = -1e30f;
#pragma unroll
      for (int n = 0; n < 4; ++n) { sacc[n][r] *= 0.125f; mx = fmaxf(mx, sacc[n][r]); }
      mx = fmaxf(mx, permxor16<1>(mx));
      mx = fmaxf(mx, permxor16<2>(mx));
      mx = fmaxf(mx, permxor16<4>(mx));
      mx = fmaxf(mx, permxor16<8>(mx));
      float mnew = fmaxf(m_i[r], mx);
      float corr = __expf(m_i[r] - mnew);
      float rs = 0.0f;
#pragma unroll
      for (int n = 0; n < 4; ++n) {
        float p = __expf(sacc[n][r] - mnew);
        sacc[n][r] = p;
        rs += p;
      }
      rs += permxor16<1>(rs);
      rs += permxor16<2>(rs);
      rs += permxor16<4>(rs);
      rs += permxor16<8>(rs);
      l_i[r] = l_i[r] * corr + rs;
      m_i[r] = mnew;
#pragma unroll
      for (int n = 0; n < 4; ++n) oacc[n][r] *= corr;
    }

    // P (C-layout) -> per-wave LDS staging (A-layout source); same-wave DS ops
    // complete in order per ISA, so no barrier is required before re-reading.
#pragma unroll
    for (int n = 0; n < 4; ++n)
#pragma unroll
      for (int r = 0; r < 8; ++r)
        pw[(r + rhalf) * 72 + n * 16 + lrow] = (bf16_t)sacc[n][r];

    // O += P @ V  via V^T tile (B-fragment K-contiguous)
#pragma unroll
    for (int ks = 0; ks < 2; ++ks) {
      v16bf pf = load_frag16(&pw[lrow * 72 + ks * 32 + khalf]);
#pragma unroll
      for (int n = 0; n < 4; ++n) {
        v16bf vf = load_frag16(&Vts[(n * 16 + lrow) * 72 + ks * 32 + khalf]);
        oacc[n] = wmma_bf16(pf, vf, oacc[n]);
      }
    }
  }

  // epilogue: O / l -> attn (bf16, [B*S, D] with head columns)
#pragma unroll
  for (int n = 0; n < 4; ++n)
#pragma unroll
    for (int r = 0; r < 8; ++r) {
      float v = oacc[n][r] / l_i[r];
      int row = i0 + wave * 16 + r + rhalf;
      attn[(size_t)(b * SEQ + row) * D_MODEL + hoff + n * 16 + lrow] = (bf16_t)v;
    }
}

// ---------------------------------------------------------------------------
extern "C" void kernel_launch(void* const* d_in, const int* in_sizes, int n_in,
                              void* d_out, int out_size, void* d_ws, size_t ws_size,
                              hipStream_t stream) {
  (void)in_sizes; (void)n_in; (void)out_size; (void)ws_size;

  const float* x     = (const float*)d_in[0];
  const float* qkv_w = (const float*)d_in[1];
  const float* qkv_b = (const float*)d_in[2];
  const float* out_w = (const float*)d_in[3];
  const float* out_b = (const float*)d_in[4];

  char* ws = (char*)d_ws;
  size_t off = 0;
  bf16_t* x_bf    = (bf16_t*)(ws + off); off += (size_t)M_ROWS * D_MODEL * 2;       //  8 MB
  bf16_t* qkvw_bf = (bf16_t*)(ws + off); off += (size_t)3 * D_MODEL * D_MODEL * 2;  //  6 MB
  bf16_t* outw_bf = (bf16_t*)(ws + off); off += (size_t)D_MODEL * D_MODEL * 2;      //  2 MB
  bf16_t* qkv_bf  = (bf16_t*)(ws + off); off += (size_t)M_ROWS * 3 * D_MODEL * 2;   // 24 MB
  bf16_t* attn_bf = (bf16_t*)(ws + off); off += (size_t)M_ROWS * D_MODEL * 2;       //  8 MB

  const int nx  = M_ROWS * D_MODEL;          // 4194304
  const int nqw = 3 * D_MODEL * D_MODEL;     // 3145728
  const int now = D_MODEL * D_MODEL;         // 1048576

  cvt_f32_bf16<<<nx  / (4 * 256), 256, 0, stream>>>(x,     x_bf,    nx);
  cvt_f32_bf16<<<nqw / (4 * 256), 256, 0, stream>>>(qkv_w, qkvw_bf, nqw);
  cvt_f32_bf16<<<now / (4 * 256), 256, 0, stream>>>(out_w, outw_bf, now);

  // qkv = x @ qkv_w^T + qkv_b   (4096 x 3072 x 1024), bf16 out
  gemm_bf16_tn<true><<<dim3(3 * D_MODEL / 128, M_ROWS / 128), 256, 0, stream>>>(
      x_bf, qkvw_bf, qkv_b, (void*)qkv_bf, M_ROWS, 3 * D_MODEL, D_MODEL);

  // attention (flash, per-(b,h,i-block))
  attention_kernel<<<dim3(SEQ / 64, BATCH * N_HEADS), 128, 0, stream>>>(qkv_bf, attn_bf);

  // out = attn @ out_w^T + out_b (4096 x 1024 x 1024), f32 out
  gemm_bf16_tn<false><<<dim3(D_MODEL / 128, M_ROWS / 128), 256, 0, stream>>>(
      attn_bf, outw_bf, out_b, d_out, M_ROWS, D_MODEL, D_MODEL);
}
